// Forest_45071386804493
// MI455X (gfx1250) — compile-verified, matrix operand order checked by reading, not run on hardware
//
#include <hip/hip_runtime.h>

#define NTREE 64
#define NLEAF 64
#define NFEAT 1024
#define NLAYER 6   // DEPTH-1

typedef float v2f  __attribute__((ext_vector_type(2)));
typedef float v8f  __attribute__((ext_vector_type(8)));
typedef _Float16 v16h __attribute__((ext_vector_type(16)));

__device__ __forceinline__ float sigmoidf(float v) {
    return 1.0f / (1.0f + __expf(-v));
}

__global__ __launch_bounds__(64)
void forest_kernel(const float* __restrict__ x,
                   const int*   __restrict__ fidx,
                   const float* __restrict__ pi,
                   float* __restrict__ out_prob,    // (B, 64)
                   float* __restrict__ out_probs)   // (B, 64 leaves, 64 trees)
{
    __shared__ float xrow[NFEAT];               // 4 KB
    __shared__ float mu_lds[NTREE * 65];        // 64x64 padded, ~16.6 KB
    __shared__ float c_lds[4 * 16 * 17];        // 4 diag tiles, padded

    const int t = threadIdx.x;                  // tree id 0..63
    const int b = blockIdx.x;                   // batch row

    // ---- 1) stage x row into LDS (coalesced float4) ----
    const float4* xg = reinterpret_cast<const float4*>(x + (size_t)b * NFEAT);
    float4* xs = reinterpret_cast<float4*>(xrow);
    #pragma unroll
    for (int i = 0; i < NFEAT / 4 / 64; ++i)    // 4 iterations
        xs[i * 64 + t] = xg[i * 64 + t];
    __syncthreads();

    // ---- 2) routing distribution mu[64] in registers, thread = tree ----
    float mu[NLEAF];
    mu[0] = 1.0f;
    const int* my_idx = fidx + t * (NLEAF - 1);
    int begin = 0;
    #pragma unroll
    for (int layer = 0; layer < NLAYER; ++layer) {
        const int n = 1 << layer;
        #pragma unroll
        for (int i = n - 1; i >= 0; --i) {
            const float s = sigmoidf(xrow[my_idx[begin + i]]);
            const float m = mu[i];
            mu[2 * i]     = m * s;
            mu[2 * i + 1] = m * (1.0f - s);
        }
        begin += n;
    }

    // ---- 3) probs[b][l][t] = mu[l]; each store = 128B contiguous per wave ----
    float* pb = out_probs + (size_t)b * (NLEAF * NTREE) + t;
    #pragma unroll
    for (int l = 0; l < NLEAF; ++l)
        pb[(size_t)l * NTREE] = mu[l];

    // ---- 4) stage mu into LDS (padded rows: bank-conflict-free column reads) ----
    #pragma unroll
    for (int l = 0; l < NLEAF; ++l)
        mu_lds[t * 65 + l] = mu[l];
    __syncthreads();

    // ---- 5) prob via WMMA: per 16-tree group g, C = Mu_g(16x64) x Pi_g^T(64x16),
    //         prob for those trees = diag(C). Wave w handles groups 2w, 2w+1. ----
    const int wave = t >> 5;
    const int lane = t & 31;
    const int lo   = (lane < 16) ? 1 : 0;
    const int ln   = lane & 15;

    #pragma unroll
    for (int gi = 0; gi < 2; ++gi) {
        const int g = 2 * wave + gi;
        const int arow = (16 * g + ln) * 65;    // mu_lds row (tree in group = M and N)
        const int brow = (16 * g + ln) * 64;    // pi row
        v8f c = {0.f, 0.f, 0.f, 0.f, 0.f, 0.f, 0.f, 0.f};

#if __has_builtin(__builtin_amdgcn_wmma_f32_16x16x4_f32)
        // f32 path: A 16x4 layout — lanes 0-15: K0,K1; lanes 16-31: K2,K3 (B symmetric)
        const int koff = lo ? 0 : 2;
        #pragma unroll
        for (int ch = 0; ch < 16; ++ch) {
            const int k = 4 * ch + koff;
            v2f a, bm;
            a[0]  = mu_lds[arow + k];
            a[1]  = mu_lds[arow + k + 1];
            bm[0] = pi[brow + k];
            bm[1] = pi[brow + k + 1];
            c = __builtin_amdgcn_wmma_f32_16x16x4_f32(false, a, false, bm,
                                                      (short)0, c, false, false);
        }
#else
        // f16 fallback: two 16x16x32 WMMAs cover K=64
        #pragma unroll
        for (int cc = 0; cc < 2; ++cc) {
            v16h a, bm;
            #pragma unroll
            for (int h = 0; h < 16; ++h) {
                const int k = 32 * cc + ((h < 8) ? ((lo ? 0 : 8) + h)
                                                 : ((lo ? 16 : 24) + (h - 8)));
                a[h]  = (_Float16)mu_lds[arow + k];
                bm[h] = (_Float16)pi[brow + k];
            }
            c = __builtin_amdgcn_wmma_f32_16x16x32_f16(false, a, false, bm,
                                                       (short)0, c, false, false);
        }
#endif
        // scatter C tile to LDS: VGPR r holds row (lo ? r : 8+r), col = ln
        float* cg = c_lds + g * (16 * 17);
        #pragma unroll
        for (int r = 0; r < 8; ++r) {
            const int row = lo ? r : (8 + r);
            cg[row * 17 + ln] = c[r];
        }
    }
    __syncthreads();

    // ---- 6) prob[b][t] = diagonal element of its group's C tile ----
    const int gg = t >> 4;
    const int ii = t & 15;
    out_prob[(size_t)b * NTREE + t] = c_lds[gg * (16 * 17) + ii * 17 + ii];
}

extern "C" void kernel_launch(void* const* d_in, const int* in_sizes, int n_in,
                              void* d_out, int out_size, void* d_ws, size_t ws_size,
                              hipStream_t stream) {
    const float* x    = (const float*)d_in[0];   // (B, 1024) f32
    const int*   fidx = (const int*)  d_in[1];   // (64, 63) i32
    const float* pi   = (const float*)d_in[2];   // (64, 64) f32
    const int B = in_sizes[0] / NFEAT;

    float* out_prob  = (float*)d_out;                        // (B, 64)
    float* out_probs = (float*)d_out + (size_t)B * NTREE;    // (B, 64, 64)

    forest_kernel<<<B, 64, 0, stream>>>(x, fidx, pi, out_prob, out_probs);
}
